// MinimalRNNCell_66786741453052
// MI455X (gfx1250) — matrix operand order repaired and checked by reading, not compiled, not run on hardware
//
#include <hip/hip_runtime.h>

#define B_ 64
#define T_ 2048
#define D_ 128
#define U_ 128

typedef __attribute__((ext_vector_type(16))) _Float16     v16h;
typedef __attribute__((ext_vector_type(8)))  float        v8f;
typedef __attribute__((ext_vector_type(4)))  float        v4f;
typedef __attribute__((ext_vector_type(4)))  unsigned int v4u;
typedef __attribute__((ext_vector_type(8)))  unsigned int v8u;

// A-matrix (16x32 f16) K index for element e of the v16h fragment.
// lanes 0-15  : K = {0..7, 16..23}   (half = 0)
// lanes 16-31 : K = {8..15, 24..31}  (half = 1)
__device__ __forceinline__ int kA(int e, int half) {
    return (e < 8) ? (half * 8 + e) : (16 + half * 8 + (e - 8));
}

// B-matrix (32x16 f16) K index for element e (contiguous order; unused, B fragments use kA):
// lanes 0-15 hold K=0..15, lanes 16-31 hold K=16..31.
__device__ __forceinline__ int kB(int e, int half) {
    return half * 16 + e;
}

__global__ __launch_bounds__(256)
void rnn_scan_wmma(const float* __restrict__ x,
                   const float* __restrict__ h0,
                   const float* __restrict__ W,
                   const float* __restrict__ R,
                   float* __restrict__ out)
{
    // Double-buffered h and x tiles (16 rows x 128 cols, f16): 16 KB total.
    __shared__ __align__(16) _Float16 hbuf[2][16 * D_];
    __shared__ __align__(16) _Float16 xbuf[2][16 * D_];

    const int tid  = threadIdx.x;
    const int lane = tid & 31;
    const int wave = tid >> 5;        // 0..7 -> owns U columns [16*wave, 16*wave+16)
    const int half = lane >> 4;       // 0 or 1
    const int ln   = lane & 15;       // N (for B/C/D) or M (for A)
    const int cb   = wave * 16;       // column base in U
    const int b0   = blockIdx.x * 16; // batch row base

    // ---- One-time: load this wave's 16-column slices of W and R as B-fragments.
    v16h wfrag[4], rfrag[4];
    #pragma unroll
    for (int kb = 0; kb < 4; ++kb) {
        v16h wf, rf;
        #pragma unroll
        for (int e = 0; e < 16; ++e) {
            const int k = kb * 32 + kA(e, half);
            wf[e] = (_Float16)W[(size_t)k * U_ + cb + ln];
            rf[e] = (_Float16)R[(size_t)k * U_ + cb + ln];
        }
        wfrag[kb] = wf;
        rfrag[kb] = rf;
    }

    // ---- Cooperative init: h0 -> hbuf[0], x_0 -> xbuf[0] (256 thr * 8 elems).
    const int row  = tid >> 4;        // 0..15
    const int colb = (tid & 15) * 8;  // 0..120
    {
        const float* hs = h0 + (size_t)(b0 + row) * U_ + colb;
        const float* xs = x  + ((size_t)(b0 + row) * T_ + 0) * D_ + colb;
        #pragma unroll
        for (int j = 0; j < 8; ++j) {
            hbuf[0][row * D_ + colb + j] = (_Float16)hs[j];
            xbuf[0][row * D_ + colb + j] = (_Float16)xs[j];
        }
    }
    __syncthreads();

    for (int t = 0; t < T_; ++t) {
        const int cur = t & 1, nxt = cur ^ 1;

        // 1) Prefetch x_{t+1} tile into registers (independent of recurrence).
        float pre[8];
        {
            const int tp = (t + 1 < T_) ? (t + 1) : t;
            const float* xs = x + ((size_t)(b0 + row) * T_ + tp) * D_ + colb;
            v4f p0 = *(const v4f*)xs;
            v4f p1 = *(const v4f*)(xs + 4);
            pre[0] = p0.x; pre[1] = p0.y; pre[2] = p0.z; pre[3] = p0.w;
            pre[4] = p1.x; pre[5] = p1.y; pre[6] = p1.z; pre[7] = p1.w;
        }

        // 2) c = x_t @ W  (x-part first: off the h critical path)
        v8f c = {0.f, 0.f, 0.f, 0.f, 0.f, 0.f, 0.f, 0.f};
        #pragma unroll
        for (int kb = 0; kb < 4; ++kb) {
            // A fragment: elements 0..7 and 8..15 are each 8 contiguous f16 -> 2x b128 LDS loads
            const _Float16* sp = &xbuf[cur][ln * D_ + kb * 32 + half * 8];
            v4u lo = *(const v4u*)sp;
            v4u hi = *(const v4u*)(sp + 16);
            v8u av; av[0]=lo[0]; av[1]=lo[1]; av[2]=lo[2]; av[3]=lo[3];
                    av[4]=hi[0]; av[5]=hi[1]; av[6]=hi[2]; av[7]=hi[3];
            v16h a = __builtin_bit_cast(v16h, av);
            c = __builtin_amdgcn_wmma_f32_16x16x32_f16(false, a, false, wfrag[kb],
                                                       (short)0, c, false, false);
        }

        //    c += h_{t-1} @ R
        #pragma unroll
        for (int kb = 0; kb < 4; ++kb) {
            const _Float16* sp = &hbuf[cur][ln * D_ + kb * 32 + half * 8];
            v4u lo = *(const v4u*)sp;
            v4u hi = *(const v4u*)(sp + 16);
            v8u av; av[0]=lo[0]; av[1]=lo[1]; av[2]=lo[2]; av[3]=lo[3];
                    av[4]=hi[0]; av[5]=hi[1]; av[6]=hi[2]; av[7]=hi[3];
            v16h a = __builtin_bit_cast(v16h, av);
            c = __builtin_amdgcn_wmma_f32_16x16x32_f16(false, a, false, rfrag[kb],
                                                       (short)0, c, false, false);
        }

        // 3) Store h_t: f32 to global output, f16 to next LDS buffer.
        #pragma unroll
        for (int r = 0; r < 8; ++r) {
            const int m = half * 8 + r;            // C/D layout: M = 8*half + r
            out[((size_t)(b0 + m) * T_ + t) * U_ + cb + ln] = c[r];
            hbuf[nxt][m * D_ + cb + ln] = (_Float16)c[r];
        }

        // 4) Commit prefetched x_{t+1} tile.
        #pragma unroll
        for (int j = 0; j < 8; ++j)
            xbuf[nxt][row * D_ + colb + j] = (_Float16)pre[j];

        __syncthreads();
    }
}

extern "C" void kernel_launch(void* const* d_in, const int* in_sizes, int n_in,
                              void* d_out, int out_size, void* d_ws, size_t ws_size,
                              hipStream_t stream) {
    (void)in_sizes; (void)n_in; (void)out_size; (void)d_ws; (void)ws_size;
    const float* x  = (const float*)d_in[0];
    const float* h0 = (const float*)d_in[1];
    const float* W  = (const float*)d_in[2];
    const float* R  = (const float*)d_in[3];
    float* out = (float*)d_out;

    rnn_scan_wmma<<<dim3(B_ / 16), dim3(256), 0, stream>>>(x, h0, W, R, out);
}
